// EncDecAD_32160715113019
// MI455X (gfx1250) — compile-verified, matrix operand order checked by reading, not run on hardware
//
#include <hip/hip_runtime.h>
#include <hip/hip_bf16.h>

// ---------------- problem constants ----------------
#define HID   1024
#define OBS   64
#define SEQ   1024
#define BATCH 128
#define FORGET_BIAS 1.0f

// ---------------- kernel geometry ----------------
#define NWG     32          // persistent workgroups; each owns 32 hidden units
#define UPG     32          // hidden units per workgroup
#define THREADS 256         // 8 wave32 waves
#define KT_ENC  34          // (OBS + HID) / 32
#define KT_DEC  64          // (HID + HID) / 32
#define KT_S    2           // OBS / 32
#define KT_O    32          // HID / 32

typedef __attribute__((ext_vector_type(16))) __bf16 v16bf;
typedef __attribute__((ext_vector_type(8)))  __bf16 v8bf;
typedef __attribute__((ext_vector_type(4)))  __bf16 v4bf;
typedef __attribute__((ext_vector_type(8)))  float  v8f;
typedef __attribute__((ext_vector_type(4)))  unsigned int u32x4;
typedef __attribute__((ext_vector_type(4)))  int i32x4;
typedef __attribute__((ext_vector_type(8)))  int i32x8;

// ---------------- workspace layout (bytes) ----------------
#define OFF_CNT  0
#define OFF_H    256                                  // [2][BATCH][HID] bf16 (double-buffered h)
#define OFF_INP  (OFF_H   + 2*BATCH*HID*2)            // [BATCH][HID] bf16 (decoder shape-linear out)
#define OFF_C    (OFF_INP + BATCH*HID*2)              // [BATCH][HID] bf16 (cell state snapshot)
#define OFF_WENC (OFF_C   + BATCH*HID*2)              // packed W_enc bf16
#define OFF_WDEC (OFF_WENC + (OBS+HID)*4*HID*2)       // packed W_dec bf16
#define OFF_WS   (OFF_WDEC + (2*HID)*4*HID*2)         // packed W_s bf16
#define OFF_WO   (OFF_WS   + OBS*HID*2)               // packed W_o bf16
#define ZERO_BYTES (OFF_H + 2*BATCH*HID*2)            // counter + h double buffer

// ---------------- TDM availability ----------------
#if defined(__AMDGCN__) && defined(__gfx1250__) && \
    __has_builtin(__builtin_amdgcn_tensor_load_to_lds) && \
    __has_builtin(__builtin_amdgcn_s_wait_tensorcnt)
#  define HAVE_TDM 1
#else
#  define HAVE_TDM 0
#endif

// compile-probe messages (visible in stderr even on success)
#define PSTR2(x) #x
#define PSTR(x) PSTR2(x)
#pragma message("probe: clang_major=" PSTR(__clang_major__))
#if HAVE_TDM
#pragma message("probe: HAVE_TDM=1")
#else
#pragma message("probe: HAVE_TDM=0")
#endif

// ================== WMMA helpers ==================
__device__ __forceinline__ v8f wmma_bf16(v16bf a, v16bf b, v8f c) {
  return __builtin_amdgcn_wmma_f32_16x16x32_bf16(false, a, false, b, (short)0, c,
                                                 false, false);
}

// A-fragment (16x32, bf16) from a row-major bf16 buffer (ISA 7.12.2 lane layout)
__device__ __forceinline__ v16bf load_a_g(const __bf16* __restrict__ base, int stride,
                                          int mrow, int k0, int lane) {
  int r  = mrow + (lane & 15);
  int kb = k0 + ((lane >> 4) << 3);
  union { v16bf v; struct { v8bf lo, hi; } s; } u;
  u.s.lo = *(const v8bf*)(base + (long)r * stride + kb);
  u.s.hi = *(const v8bf*)(base + (long)r * stride + kb + 16);
  return u.v;
}

// B-fragment from prepacked tile (512 bf16, lane-major: [lane][16])
__device__ __forceinline__ v16bf load_b_t(const __bf16* __restrict__ tile, int lane) {
  return *(const v16bf*)(tile + lane * 16);
}

__device__ __forceinline__ float sigmoidf_(float x) {
  return 1.0f / (1.0f + __expf(-x));
}
__device__ __forceinline__ float tanhf_(float x) {
  float e = __expf(2.0f * x);
  return (e - 1.0f) / (e + 1.0f);
}

// device-wide barrier: monotonic counter in ws
__device__ __forceinline__ void grid_barrier(unsigned* cnt, unsigned& barno) {
  __syncthreads();
  barno++;
  if (threadIdx.x == 0) {
    __threadfence();
    atomicAdd(cnt, 1u);
    const unsigned tgt = barno * NWG;
    while (__hip_atomic_load(cnt, __ATOMIC_RELAXED, __HIP_MEMORY_SCOPE_AGENT) < tgt)
      __builtin_amdgcn_s_sleep(2);
    __threadfence();
  }
  __syncthreads();
}

// ================== TDM: DMA one x_t tile (BATCH*OBS f32) into LDS ==================
#if HAVE_TDM
__device__ __forceinline__ void tdm_load_x(const float* gsrc, float* lds_dst) {
  unsigned long long ga = (unsigned long long)gsrc;
  // NOTE: LDS offset taken from flat address (descriptor semantics unchecked here)
  unsigned lds = (unsigned)(unsigned long long)lds_dst;
  u32x4 g0;
  g0.x = 1u;                                            // count=1, user descriptor
  g0.y = lds;                                           // lds_addr
  g0.z = (unsigned)ga;                                  // global_addr[31:0]
  g0.w = (unsigned)((ga >> 32) & 0x01FFFFFFull) | (2u << 30);  // addr[56:32] | type=2
  i32x8 g1;
  g1[0] = (2 << 16);                 // data_size = 4 bytes
  g1[1] = (int)(8192u << 16);        // tensor_dim0[15:0] = 8192
  g1[2] = (1 << 16);                 // tensor_dim0 hi=0 | tensor_dim1[15:0]=1
  g1[3] = (int)(8192u << 16);        // tensor_dim1 hi=0 | tile_dim0 = 8192
  g1[4] = 1;                         // tile_dim1 = 1, tile_dim2 = 0
  g1[5] = 8192;                      // tensor_dim0_stride
  g1[6] = 0;
  g1[7] = 0;
  i32x4 z4 = {0, 0, 0, 0};
#if defined(__clang_major__) && __clang_major__ >= 23
  i32x8 z8 = {0, 0, 0, 0, 0, 0, 0, 0};
  __builtin_amdgcn_tensor_load_to_lds(g0, g1, z4, z4, z8, 0);
#else
  __builtin_amdgcn_tensor_load_to_lds(g0, g1, z4, z4, 0);
#endif
}
#endif

// split-phase x staging ------------------------------------------------------
// x_issue: start async DMA (or prefetch) of the next needed x tile
__device__ __forceinline__ void x_issue(const float* xt, float* xf, int tid) {
#if HAVE_TDM
  if (tid < 32) tdm_load_x(xt, xf);
#else
  (void)xf;
  __builtin_prefetch(xt + tid * 32, 0, 1);
#endif
}

// x_convert: wait for DMA, convert staged f32 -> bf16 for WMMA A-fragments
__device__ __forceinline__ void x_convert(const float* xt, float* xf, __bf16* xbf,
                                          int tid) {
#if HAVE_TDM
  (void)xt;
  if (tid < 32) __builtin_amdgcn_s_wait_tensorcnt(0);
  __syncthreads();
  for (int i = tid * 4; i < BATCH * OBS; i += THREADS * 4) {
    float4 v = *(const float4*)(xf + i);
    v4bf b;
    b.x = (__bf16)v.x; b.y = (__bf16)v.y; b.z = (__bf16)v.z; b.w = (__bf16)v.w;
    *(v4bf*)(xbf + i) = b;
  }
#else
  (void)xf;
  for (int i = tid; i < BATCH * OBS; i += THREADS)
    xbf[i] = (__bf16)xt[i];
#endif
  __syncthreads();
}

// ================== weight prepacking ==================
// dst layout: [g][kt][nt][lane(32)][j(16)]; element = W[k(kt,lane,j)][col(g,nt,lane)]
// mode 0: gate-grouped (W_enc/W_dec): NT=8, col = (nt>>1)*HID + g*UPG + (nt&1)*16 + n
// mode 1: W_s:                        NT=2, col = g*UPG + nt*16 + n
// mode 2: W_o:                        NT=4, col = nt*16 + n (g == 0)
__global__ void pack_kernel(const float* __restrict__ W, __bf16* __restrict__ dst,
                            int KT, int NT, int ld, int mode, int total) {
  for (int idx = blockIdx.x * blockDim.x + threadIdx.x; idx < total;
       idx += gridDim.x * blockDim.x) {
    int j    = idx & 15;
    int l    = (idx >> 4) & 31;
    int rest = idx >> 9;
    int nt   = rest % NT;
    int kt   = (rest / NT) % KT;
    int g    = (rest / NT) / KT;
    int n    = l & 15;
    int kb   = (l >> 4) << 3;
    int k    = kt * 32 + kb + ((j < 8) ? j : (j + 8));
    int col;
    if (mode == 0)      col = (nt >> 1) * HID + g * UPG + (nt & 1) * 16 + n;
    else if (mode == 1) col = g * UPG + nt * 16 + n;
    else                col = nt * 16 + n;
    dst[idx] = (__bf16)W[(long)k * ld + col];
  }
}

// ================== out_linear: [B,HID](bf16) @ Wo -> [B,OBS](f32) ==================
__device__ __forceinline__ void out_linear(const __bf16* __restrict__ cbuf,
                                           const __bf16* __restrict__ Wop,
                                           float* __restrict__ dst,
                                           const float* bol, int mbase, int rofs,
                                           int lane) {
  v8f acc[4] = {};
  for (int kt = 0; kt < KT_O; ++kt) {
    v16bf a = load_a_g(cbuf, HID, mbase, kt * 32, lane);
    const __bf16* bt = Wop + (long)kt * 4 * 512;
#pragma unroll
    for (int nt = 0; nt < 4; ++nt)
      acc[nt] = wmma_bf16(a, load_b_t(bt + nt * 512, lane), acc[nt]);
  }
#pragma unroll
  for (int nt = 0; nt < 4; ++nt) {
    int col = nt * 16 + (lane & 15);
#pragma unroll
    for (int r = 0; r < 8; ++r)
      dst[(long)(rofs + r) * OBS + col] = acc[nt][r] + bol[nt];
  }
}

// ================== persistent encoder/decoder kernel ==================
__global__ __launch_bounds__(THREADS)
void encdec_persistent(const float* __restrict__ xin,   // [SEQ][BATCH][OBS]
                       const float* __restrict__ b_enc, const float* __restrict__ b_dec,
                       const float* __restrict__ b_s,   const float* __restrict__ b_o,
                       unsigned char* __restrict__ ws,
                       float* __restrict__ out) {       // [SEQ][BATCH][OBS]
  unsigned*     cnt    = (unsigned*)(ws + OFF_CNT);
  __bf16*       hbuf   = (__bf16*)(ws + OFF_H);
  __bf16*       inpbuf = (__bf16*)(ws + OFF_INP);
  __bf16*       cbuf   = (__bf16*)(ws + OFF_C);
  const __bf16* Wenc   = (const __bf16*)(ws + OFF_WENC);
  const __bf16* Wdec   = (const __bf16*)(ws + OFF_WDEC);
  const __bf16* Wsp    = (const __bf16*)(ws + OFF_WS);
  const __bf16* Wop    = (const __bf16*)(ws + OFF_WO);

  const int g     = blockIdx.x;
  const int lane  = threadIdx.x & 31;
  const int wave  = threadIdx.x >> 5;
  const int mbase = wave * 16;
  const int rofs  = mbase + ((lane >> 4) << 3);   // C/D layout: row = rofs + r

  __shared__ __align__(16) float  xf_s[BATCH * OBS];  // 32 KB: raw f32 x_t (TDM target)
  __shared__ __align__(32) __bf16 x_s[BATCH * OBS];   // 16 KB: bf16 x_t

  // ---- biases resident in registers for the whole run ----
  float bei[2], bej[2], bef[2], beo[2];
  float bdi[2], bdj[2], bdf[2], bdo[2];
  float bss[2];
#pragma unroll
  for (int p2 = 0; p2 < 2; ++p2) {
    int u = g * UPG + p2 * 16 + (lane & 15);
    bei[p2] = b_enc[u];           bej[p2] = b_enc[HID + u];
    bef[p2] = b_enc[2 * HID + u] + FORGET_BIAS;
    beo[p2] = b_enc[3 * HID + u];
    bdi[p2] = b_dec[u];           bdj[p2] = b_dec[HID + u];
    bdf[p2] = b_dec[2 * HID + u] + FORGET_BIAS;
    bdo[p2] = b_dec[3 * HID + u];
    bss[p2] = b_s[u];
  }
  float bol[4];
#pragma unroll
  for (int nt = 0; nt < 4; ++nt) bol[nt] = b_o[nt * 16 + (lane & 15)];

  // ---- cell state lives in VGPRs across all 2047 steps ----
  float cst[2][8];
#pragma unroll
  for (int p2 = 0; p2 < 2; ++p2)
#pragma unroll
    for (int r = 0; r < 8; ++r) cst[p2][r] = 0.f;

  unsigned barno = 0;
  int p = 0;   // h double-buffer parity (hbuf[p] holds current h; initially zeros)

#define LOAD_A_ENC(kt) ((kt) < 2 ? load_a_g(x_s, OBS, mbase, (kt) * 32, lane) \
                                 : load_a_g(hread, HID, mbase, (kt) * 32 - OBS, lane))
#define LOAD_A_DEC(kt) ((kt) < 32 ? load_a_g(inpbuf, HID, mbase, (kt) * 32, lane) \
                                  : load_a_g(hread, HID, mbase, ((kt) - 32) * 32, lane))

  // kick off DMA of the first x tile
  x_issue(xin, xf_s, threadIdx.x);

  // ======== encoder: t = 0 .. SEQ-1 ========
  for (int t = 0; t < SEQ; ++t) {
    x_convert(xin + (long)t * BATCH * OBS, xf_s, x_s, threadIdx.x);
    // start DMA of next needed x (x[t+1]; at t=SEQ-1 the decoder's first x = x[SEQ-1])
    {
      int tn = (t + 1 < SEQ) ? (t + 1) : (SEQ - 1);
      x_issue(xin + (long)tn * BATCH * OBS, xf_s, threadIdx.x);
    }

    const __bf16* hread = hbuf + (long)p * BATCH * HID;
    const __bf16* wbase = Wenc + (long)g * KT_ENC * 8 * 512;
    v8f acc[8] = {};
    v16bf b0[4], b1[4];
#pragma unroll
    for (int i = 0; i < 4; ++i) b0[i] = load_b_t(wbase + i * 512, lane);
    v16bf a_cur = LOAD_A_ENC(0);

    for (int kt = 0; kt < KT_ENC; ++kt) {
      const __bf16* bt = wbase + (long)kt * 8 * 512;
#pragma unroll
      for (int i = 0; i < 4; ++i) b1[i] = load_b_t(bt + (4 + i) * 512, lane);
      if (kt + 2 < KT_ENC)
        __builtin_prefetch(wbase + (long)(kt + 2) * 8 * 512 + lane * 128, 0, 3);
      v16bf a_next = a_cur;
      if (kt + 1 < KT_ENC) a_next = LOAD_A_ENC(kt + 1);
#pragma unroll
      for (int nt = 0; nt < 4; ++nt) acc[nt] = wmma_bf16(a_cur, b0[nt], acc[nt]);
      if (kt + 1 < KT_ENC) {
        const __bf16* btn = wbase + (long)(kt + 1) * 8 * 512;
#pragma unroll
        for (int i = 0; i < 4; ++i) b0[i] = load_b_t(btn + i * 512, lane);
      }
#pragma unroll
      for (int nt = 4; nt < 8; ++nt) acc[nt] = wmma_bf16(a_cur, b1[nt - 4], acc[nt]);
      a_cur = a_next;
    }

    __bf16* hwrite = hbuf + (long)(p ^ 1) * BATCH * HID;
    const bool last = (t == SEQ - 1);
#pragma unroll
    for (int pp = 0; pp < 2; ++pp) {
      int unit = g * UPG + pp * 16 + (lane & 15);
#pragma unroll
      for (int r = 0; r < 8; ++r) {
        float iv = acc[0 + pp][r] + bei[pp];
        float jv = acc[2 + pp][r] + bej[pp];
        float fv = acc[4 + pp][r] + bef[pp];
        float ov = acc[6 + pp][r] + beo[pp];
        float cn = sigmoidf_(fv) * cst[pp][r] + sigmoidf_(iv) * tanhf_(jv);
        float hn = sigmoidf_(ov) * tanhf_(cn);
        cst[pp][r] = cn;
        int row = rofs + r;
        hwrite[(long)row * HID + unit] = (__bf16)hn;
        if (last) cbuf[(long)row * HID + unit] = (__bf16)cn;
      }
    }
    p ^= 1;
    grid_barrier(cnt, barno);
  }

  // out_init = c_f @ W_o + b_o -> outputs[SEQ-1]
  if (g == 0)
    out_linear(cbuf, Wop, out + (long)(SEQ - 1) * BATCH * OBS, bol, mbase, rofs, lane);

  // ======== decoder: k = 0 .. SEQ-2 (input = input_seq[SEQ-1-k]) ========
  for (int k = 0; k < SEQ - 1; ++k) {
    // --- phase 1: inp = tanh(x @ W_s + b_s), each WG its 32-unit slice ---
    x_convert(xin + (long)(SEQ - 1 - k) * BATCH * OBS, xf_s, x_s, threadIdx.x);

    v8f sacc[2] = {};
    for (int kt = 0; kt < KT_S; ++kt) {
      v16bf a = load_a_g(x_s, OBS, mbase, kt * 32, lane);
      const __bf16* bt = Wsp + (long)((g * KT_S + kt) * 2) * 512;
#pragma unroll
      for (int nt = 0; nt < 2; ++nt)
        sacc[nt] = wmma_bf16(a, load_b_t(bt + nt * 512, lane), sacc[nt]);
    }
#pragma unroll
    for (int pp = 0; pp < 2; ++pp) {
      int unit = g * UPG + pp * 16 + (lane & 15);
#pragma unroll
      for (int r = 0; r < 8; ++r)
        inpbuf[(long)(rofs + r) * HID + unit] = (__bf16)tanhf_(sacc[pp][r] + bss[pp]);
    }
    // start DMA of the next decoder x (x[SEQ-2-k]) -- overlaps phase 2 + barriers
    if (k + 1 < SEQ - 1)
      x_issue(xin + (long)(SEQ - 2 - k) * BATCH * OBS, xf_s, threadIdx.x);
    grid_barrier(cnt, barno);

    // --- phase 2: z = [inp, h] @ W_dec + b_dec; LSTM cell update ---
    const __bf16* hread = hbuf + (long)p * BATCH * HID;
    const __bf16* wbase = Wdec + (long)g * KT_DEC * 8 * 512;
    v8f acc[8] = {};
    v16bf b0[4], b1[4];
#pragma unroll
    for (int i = 0; i < 4; ++i) b0[i] = load_b_t(wbase + i * 512, lane);
    v16bf a_cur = LOAD_A_DEC(0);

    for (int kt = 0; kt < KT_DEC; ++kt) {
      const __bf16* bt = wbase + (long)kt * 8 * 512;
#pragma unroll
      for (int i = 0; i < 4; ++i) b1[i] = load_b_t(bt + (4 + i) * 512, lane);
      if (kt + 2 < KT_DEC)
        __builtin_prefetch(wbase + (long)(kt + 2) * 8 * 512 + lane * 128, 0, 3);
      v16bf a_next = a_cur;
      if (kt + 1 < KT_DEC) a_next = LOAD_A_DEC(kt + 1);
#pragma unroll
      for (int nt = 0; nt < 4; ++nt) acc[nt] = wmma_bf16(a_cur, b0[nt], acc[nt]);
      if (kt + 1 < KT_DEC) {
        const __bf16* btn = wbase + (long)(kt + 1) * 8 * 512;
#pragma unroll
        for (int i = 0; i < 4; ++i) b0[i] = load_b_t(btn + i * 512, lane);
      }
#pragma unroll
      for (int nt = 4; nt < 8; ++nt) acc[nt] = wmma_bf16(a_cur, b1[nt - 4], acc[nt]);
      a_cur = a_next;
    }

    __bf16* hwrite = hbuf + (long)(p ^ 1) * BATCH * HID;
#pragma unroll
    for (int pp = 0; pp < 2; ++pp) {
      int unit = g * UPG + pp * 16 + (lane & 15);
#pragma unroll
      for (int r = 0; r < 8; ++r) {
        float iv = acc[0 + pp][r] + bdi[pp];
        float jv = acc[2 + pp][r] + bdj[pp];
        float fv = acc[4 + pp][r] + bdf[pp];
        float ov = acc[6 + pp][r] + bdo[pp];
        float cn = sigmoidf_(fv) * cst[pp][r] + sigmoidf_(iv) * tanhf_(jv);
        float hn = sigmoidf_(ov) * tanhf_(cn);
        cst[pp][r] = cn;
        int row = rofs + r;
        hwrite[(long)row * HID + unit] = (__bf16)hn;
        cbuf[(long)row * HID + unit]   = (__bf16)cn;
      }
    }
    p ^= 1;
    grid_barrier(cnt, barno);

    // --- phase 3: y_k = c @ W_o + b_o -> outputs[SEQ-2-k] (WG0 only) ---
    if (g == 0)
      out_linear(cbuf, Wop, out + (long)(SEQ - 2 - k) * BATCH * OBS, bol, mbase, rofs, lane);
  }
#undef LOAD_A_ENC
#undef LOAD_A_DEC
}

// ================== host entry ==================
extern "C" void kernel_launch(void* const* d_in, const int* in_sizes, int n_in,
                              void* d_out, int out_size, void* d_ws, size_t ws_size,
                              hipStream_t stream) {
  (void)in_sizes; (void)n_in; (void)out_size; (void)ws_size;
  const float* input_seq = (const float*)d_in[0];
  const float* W_enc     = (const float*)d_in[1];
  const float* b_enc     = (const float*)d_in[2];
  const float* W_dec     = (const float*)d_in[3];
  const float* b_dec     = (const float*)d_in[4];
  const float* W_s       = (const float*)d_in[5];
  const float* b_s       = (const float*)d_in[6];
  const float* W_o       = (const float*)d_in[7];
  const float* b_o       = (const float*)d_in[8];
  unsigned char* ws = (unsigned char*)d_ws;
  float* out = (float*)d_out;

  // zero the barrier counter + h double-buffer (initial state h=0)
  hipMemsetAsync(ws, 0, ZERO_BYTES, stream);

  pack_kernel<<<512, 256, 0, stream>>>(W_enc, (__bf16*)(ws + OFF_WENC),
                                       KT_ENC, 8, 4 * HID, 0, NWG * KT_ENC * 8 * 512);
  pack_kernel<<<512, 256, 0, stream>>>(W_dec, (__bf16*)(ws + OFF_WDEC),
                                       KT_DEC, 8, 4 * HID, 0, NWG * KT_DEC * 8 * 512);
  pack_kernel<<<64, 256, 0, stream>>>(W_s, (__bf16*)(ws + OFF_WS),
                                      KT_S, 2, HID, 1, NWG * KT_S * 2 * 512);
  pack_kernel<<<64, 256, 0, stream>>>(W_o, (__bf16*)(ws + OFF_WO),
                                      KT_O, 4, OBS, 2, KT_O * 4 * 512);

  encdec_persistent<<<NWG, THREADS, 0, stream>>>(input_seq, b_enc, b_dec, b_s, b_o,
                                                 ws, out);
}
